// TimeMixModule_63299228008938
// MI455X (gfx1250) — matrix operand order repaired and checked by reading, not compiled, not run on hardware
//
#include <hip/hip_runtime.h>
#include <hip/hip_bf16.h>
#include <cstdint>
#include <cstddef>

// ---------------- problem constants ----------------
#define NB 4
#define TT 2048
#define CC 1024
#define MM (NB * TT)          // 8192 rows
#define CH 64                 // WKV scan chunk length
#define NCH (TT / CH)         // 32 chunks

// _EPS = exp(-LAYER_ID/N_LAYERS) = exp(-1/12)
#define EPS_BASE 0.9200444146293233f
#define EPS_R_C  (EPS_BASE * 0.5f)
#define EPS_K_C  (EPS_BASE + 0.3f * (1.0f / 11.0f))
#define EPS_V_C  (EPS_BASE)

typedef __attribute__((ext_vector_type(16))) _Float16 v16h;
typedef __attribute__((ext_vector_type(8)))  float    v8f;

// ---------------- CDNA5 async global->LDS copy ----------------
#if defined(__HIP_DEVICE_COMPILE__) && __has_builtin(__builtin_amdgcn_global_load_async_to_lds_b128)
#define HAVE_ASYNC_LDS 1
#else
#define HAVE_ASYNC_LDS 0
#endif

typedef int v4i_gcc __attribute__((vector_size(16)));
typedef __attribute__((address_space(1))) v4i_gcc* as1_v4i;
typedef __attribute__((address_space(3))) v4i_gcc* as3_v4i;

__device__ __forceinline__ void g2l_b128(const void* gp, void* lp) {
#if HAVE_ASYNC_LDS
    __builtin_amdgcn_global_load_async_to_lds_b128((as1_v4i)gp, (as3_v4i)lp, 0, 0);
#else
    *(uint4*)lp = *(const uint4*)gp;          // fallback: sync copy through VGPRs
#endif
}

__device__ __forceinline__ void wait_async_all() {
#if HAVE_ASYNC_LDS
#if __has_builtin(__builtin_amdgcn_s_wait_asynccnt)
    __builtin_amdgcn_s_wait_asynccnt(0);
#else
    asm volatile("s_wait_asynccnt 0x0" ::: "memory");
#endif
#endif
}

__device__ __forceinline__ float sigmoidf_fast(float x) {
    return 1.0f / (1.0f + __expf(-x));
}
__device__ __forceinline__ float clip_arg(float x) {
    return fminf(fmaxf(x, -20.0f), 10.0f);
}

// ---------------- kernel 1: time-interp + f32->f16, all three eps ----------------
__global__ __launch_bounds__(256)
void prep_x_kernel(const float* __restrict__ x,
                   _Float16* __restrict__ xr,
                   _Float16* __restrict__ xk,
                   _Float16* __restrict__ xv)
{
    int idx = blockIdx.x * blockDim.x + threadIdx.x;     // over N*T*C = 8388608
    if (idx >= NB * TT * CC) return;
    int t = (idx / CC) % TT;
    float prev = (t > 0)      ? x[idx - CC] : 0.0f;      // x[t-1]
    float next = (t < TT - 1) ? x[idx + CC] : 0.0f;      // x[t+1]
    xr[idx] = (_Float16)(EPS_R_C * prev + (1.0f - EPS_R_C) * next);
    xk[idx] = (_Float16)(EPS_K_C * prev + (1.0f - EPS_K_C) * next);
    xv[idx] = (_Float16)(EPS_V_C * prev + (1.0f - EPS_V_C) * next);
}

// ---------------- kernel 2: f32 -> f16 convert (weights) ----------------
__global__ __launch_bounds__(256)
void cvt_f32_f16_kernel(const float* __restrict__ src, _Float16* __restrict__ dst, int n)
{
    int idx = blockIdx.x * blockDim.x + threadIdx.x;
    if (idx < n) dst[idx] = (_Float16)src[idx];
}

// ---------------- kernel 3: tiled WMMA GEMM  Y = A(MxK) * B(NxK)^T + bias ----------------
// Block: 256 threads (8 waves) in a 4x2 wave grid. Block tile 128x128, K-step 32.
// Wave tile: 32(M) x 64(N) = 2x4 WMMA tiles -> A/B fragments reused, 12 ds_load_b128
// and a single DScnt drain per K-step, then 8 back-to-back WMMAs.
// Double-buffered LDS; async global->LDS staging overlapped with compute.
__global__ __launch_bounds__(256)
void gemm_f16_wmma_kernel(const _Float16* __restrict__ A,
                          const _Float16* __restrict__ B,
                          const float* __restrict__ bias,
                          float* __restrict__ Y,
                          int M, int Nn, int K)
{
    constexpr int ROWB  = 80;        // LDS row stride (64B payload + 16B pad, conflict-free)
    constexpr int TILEB = 128 * ROWB;     // 10240 B per matrix tile
    constexpr int BUFB  = 2 * TILEB;      // A tile + B tile per buffer
    __shared__ __align__(16) unsigned char lds[2 * BUFB];   // double buffer, 40 KB

    const int tid  = threadIdx.x;
    const int wave = tid >> 5;
    const int lane = tid & 31;
    const int waveRow = wave >> 1;         // 0..3  -> 32 rows each
    const int waveCol = wave & 1;          // 0..1  -> 64 cols each

    const int m0 = blockIdx.x * 128;
    const int n0 = blockIdx.y * 128;

    // cooperative staging: 4 threads per 64B row-slice, one b128 each
    const int copyRow = tid >> 2;          // 0..63
    const int copyOff = (tid & 3) * 16;    // 0,16,32,48 bytes

    // WMMA fragment addressing (ISA 7.12.2, 16-bit, wave32)
    const int fragRow = lane & 15;
    const int aKlo    = (lane >= 16) ? 16 : 0;   // A: lane-half -> k+8 (byte 16)
    const int bKlo    = (lane >= 16) ? 16 : 0;   // B: lane-half -> k+8 (byte 16)

    v8f acc[8];                            // [rt][ct] = acc[rt*4+ct]
#pragma unroll
    for (int i = 0; i < 8; ++i) { v8f z = {}; acc[i] = z; }

    const int nk = K / 32;

    // ---- stage tile 0 into buffer 0 ----
    {
        unsigned char* buf = lds;
#pragma unroll
        for (int it = 0; it < 2; ++it) {
            int r = copyRow + it * 64;
            const unsigned char* ga = (const unsigned char*)A + ((size_t)(m0 + r) * K) * 2 + copyOff;
            const unsigned char* gb = (const unsigned char*)B + ((size_t)(n0 + r) * K) * 2 + copyOff;
            g2l_b128(ga, buf + r * ROWB + copyOff);
            g2l_b128(gb, buf + TILEB + r * ROWB + copyOff);
        }
    }

    for (int i = 0; i < nk; ++i) {
        wait_async_all();          // this wave's copies of tile i are done
        __syncthreads();           // everyone's copies done; prior reads of other buf done

        if (i + 1 < nk) {          // overlap: stage tile i+1 into the other buffer
            unsigned char* buf = lds + ((i + 1) & 1) * BUFB;
            int kt = (i + 1) * 32;
#pragma unroll
            for (int it = 0; it < 2; ++it) {
                int r = copyRow + it * 64;
                const unsigned char* ga = (const unsigned char*)A + ((size_t)(m0 + r) * K + kt) * 2 + copyOff;
                const unsigned char* gb = (const unsigned char*)B + ((size_t)(n0 + r) * K + kt) * 2 + copyOff;
                g2l_b128(ga, buf + r * ROWB + copyOff);
                g2l_b128(gb, buf + TILEB + r * ROWB + copyOff);
            }
        }

        // ---- compute on buffer i&1 ----
        unsigned char* ldsA = lds + (i & 1) * BUFB;
        unsigned char* ldsB = ldsA + TILEB;

        union F { uint4 q[2]; v16h v; };
        F afrag[2], bfrag[4];
#pragma unroll
        for (int rt = 0; rt < 2; ++rt) {
            const unsigned char* ap = ldsA + (waveRow * 32 + rt * 16 + fragRow) * ROWB;
            afrag[rt].q[0] = *(const uint4*)(ap + aKlo);        // k = klo+0..7
            afrag[rt].q[1] = *(const uint4*)(ap + 32 + aKlo);   // k = klo+16..23
        }
#pragma unroll
        for (int ct = 0; ct < 4; ++ct) {
            const unsigned char* bp = ldsB + (waveCol * 64 + ct * 16 + fragRow) * ROWB + bKlo;
            bfrag[ct].q[0] = *(const uint4*)(bp);
            bfrag[ct].q[1] = *(const uint4*)(bp + 32);
        }
#pragma unroll
        for (int rt = 0; rt < 2; ++rt) {
#pragma unroll
            for (int ct = 0; ct < 4; ++ct) {                    // 8 back-to-back WMMAs
                acc[rt * 4 + ct] = __builtin_amdgcn_wmma_f32_16x16x32_f16(
                    false, afrag[rt].v, false, bfrag[ct].v, (short)0,
                    acc[rt * 4 + ct], false, false);
            }
        }
    }

    // store C/D: VGPR v -> row (v + (lane>=16 ? 8 : 0)), lane&15 -> col
    const int mbase = m0 + waveRow * 32 + ((lane >= 16) ? 8 : 0);
    const int nbase = n0 + waveCol * 64 + (lane & 15);
#pragma unroll
    for (int rt = 0; rt < 2; ++rt) {
#pragma unroll
        for (int ct = 0; ct < 4; ++ct) {
            int n = nbase + ct * 16;
            float bb = bias[n];
#pragma unroll
            for (int v = 0; v < 8; ++v) {
                int m = mbase + rt * 16 + v;
                Y[(size_t)m * Nn + n] = acc[rt * 4 + ct][v] + bb;
            }
        }
    }
}

// ---------------- WKV chunked scan (3 passes) ----------------
// pass 1: per-(batch,chunk,channel) partial sums of e and e*v
__global__ __launch_bounds__(256)
void wkv_pass1_kernel(const float* __restrict__ K, const float* __restrict__ V,
                      const float* __restrict__ time_decay,
                      float* __restrict__ Pe, float* __restrict__ Pev)
{
    int idx = blockIdx.x * blockDim.x + threadIdx.x;   // NB*NCH*CC = 131072
    if (idx >= NB * NCH * CC) return;
    int c = idx % CC;
    int j = (idx / CC) % NCH;
    int n = idx / (CC * NCH);

    const float w = fmaxf(time_decay[c], 0.0f);
    const size_t base = (size_t)n * TT * CC + (size_t)j * CH * CC + c;

    float ae = 0.0f, aev = 0.0f;
    for (int tt = 0; tt < CH; ++tt) {
        int t = j * CH + tt;
        float kk = K[base + (size_t)tt * CC];
        float vv = V[base + (size_t)tt * CC];
        float e  = __expf(clip_arg(-w * (float)(TT - 1 - t) + kk));
        ae  += e;
        aev += e * vv;
    }
    Pe[idx]  = ae;     // layout [n][j][c]
    Pev[idx] = aev;
}

// pass 2: exclusive scan over the 32 chunk partials, in place
__global__ __launch_bounds__(256)
void wkv_pass2_kernel(float* __restrict__ Pe, float* __restrict__ Pev)
{
    int idx = blockIdx.x * blockDim.x + threadIdx.x;   // NB*CC = 4096
    if (idx >= NB * CC) return;
    int c = idx % CC;
    int n = idx / CC;
    float re = 0.0f, rev = 0.0f;
    for (int j = 0; j < NCH; ++j) {
        size_t o = ((size_t)n * NCH + j) * CC + c;
        float te = Pe[o], tev = Pev[o];
        Pe[o]  = re;  Pev[o] = rev;
        re  += te;  rev += tev;
    }
}

// pass 3: recompute within chunk with carried prefix; emit gated f16 output
__global__ __launch_bounds__(256)
void wkv_pass3_kernel(const float* __restrict__ R, const float* __restrict__ K,
                      const float* __restrict__ V,
                      const float* __restrict__ time_decay, const float* __restrict__ U,
                      const float* __restrict__ Pe, const float* __restrict__ Pev,
                      _Float16* __restrict__ G)
{
    int idx = blockIdx.x * blockDim.x + threadIdx.x;   // NB*NCH*CC
    if (idx >= NB * NCH * CC) return;
    int c = idx % CC;
    int j = (idx / CC) % NCH;
    int n = idx / (CC * NCH);

    const float w = fmaxf(time_decay[c], 0.0f);
    const float u = U[c];
    const size_t base = (size_t)n * TT * CC + (size_t)j * CH * CC + c;

    float acc_e  = Pe[idx];     // sums over all t < j*CH
    float acc_ev = Pev[idx];

    for (int tt = 0; tt < CH; ++tt) {
        int t = j * CH + tt;
        const size_t off = base + (size_t)tt * CC;
        const float kk = K[off];
        const float vv = V[off];
        const float rr = R[off];

        const float e  = __expf(clip_arg(-w * (float)(TT - 1 - t) + kk));
        const float eu = __expf(clip_arg(u + kk));

        const float Anum = acc_ev + eu * vv;
        const float Bden = acc_e  + eu;
        G[off] = (_Float16)(sigmoidf_fast(rr) * (Anum / Bden));

        acc_ev += e * vv;
        acc_e  += e;
    }
}

// ---------------- launch ----------------
extern "C" void kernel_launch(void* const* d_in, const int* in_sizes, int n_in,
                              void* d_out, int out_size, void* d_ws, size_t ws_size,
                              hipStream_t stream)
{
    const float* x  = (const float*)d_in[0];
    const float* Wr = (const float*)d_in[1];
    const float* br = (const float*)d_in[2];
    const float* Wk = (const float*)d_in[3];
    const float* bk = (const float*)d_in[4];
    const float* Wv = (const float*)d_in[5];
    const float* bv = (const float*)d_in[6];
    const float* Wo = (const float*)d_in[7];
    const float* bo = (const float*)d_in[8];
    const float* td = (const float*)d_in[9];
    const float* U  = (const float*)d_in[10];

    constexpr size_t SZ_X16 = (size_t)MM * CC * 2;   // 16 MB
    constexpr size_t SZ_W16 = (size_t)CC * CC * 2;   //  2 MB
    constexpr size_t SZ_F32 = (size_t)MM * CC * 4;   // 32 MB
    constexpr size_t SZ_P   = (size_t)NB * NCH * CC * 4;  // 512 KB

    char* ws = (char*)d_ws;
    _Float16* Xr  = (_Float16*)(ws);
    _Float16* Xk  = (_Float16*)(ws + SZ_X16);
    _Float16* Xv  = (_Float16*)(ws + 2 * SZ_X16);
    _Float16* Wrh = (_Float16*)(ws + 3 * SZ_X16);
    _Float16* Wkh = (_Float16*)(ws + 3 * SZ_X16 + SZ_W16);
    _Float16* Wvh = (_Float16*)(ws + 3 * SZ_X16 + 2 * SZ_W16);
    _Float16* Woh = (_Float16*)(ws + 3 * SZ_X16 + 3 * SZ_W16);
    float*    Rb  = (float*)(ws + 3 * SZ_X16 + 4 * SZ_W16);
    float*    Kb  = (float*)(ws + 3 * SZ_X16 + 4 * SZ_W16 + SZ_F32);
    float*    Vb  = (float*)(ws + 3 * SZ_X16 + 4 * SZ_W16 + 2 * SZ_F32);
    _Float16* Gb  = (_Float16*)(ws + 3 * SZ_X16 + 4 * SZ_W16 + 3 * SZ_F32);
    float*    Pe  = (float*)(ws + 4 * SZ_X16 + 4 * SZ_W16 + 3 * SZ_F32);
    float*    Pev = (float*)(ws + 4 * SZ_X16 + 4 * SZ_W16 + 3 * SZ_F32 + SZ_P);

    {   // 1) time-interp + f16 cast
        int total = NB * TT * CC;
        prep_x_kernel<<<(total + 255) / 256, 256, 0, stream>>>(x, Xr, Xk, Xv);
    }
    {   // 2) weight conversion
        int nW = CC * CC;
        int g = (nW + 255) / 256;
        cvt_f32_f16_kernel<<<g, 256, 0, stream>>>(Wr, Wrh, nW);
        cvt_f32_f16_kernel<<<g, 256, 0, stream>>>(Wk, Wkh, nW);
        cvt_f32_f16_kernel<<<g, 256, 0, stream>>>(Wv, Wvh, nW);
        cvt_f32_f16_kernel<<<g, 256, 0, stream>>>(Wo, Woh, nW);
    }
    {   // 3) R/K/V projection GEMMs
        dim3 grid(MM / 128, CC / 128);
        gemm_f16_wmma_kernel<<<grid, 256, 0, stream>>>(Xr, Wrh, br, Rb, MM, CC, CC);
        gemm_f16_wmma_kernel<<<grid, 256, 0, stream>>>(Xk, Wkh, bk, Kb, MM, CC, CC);
        gemm_f16_wmma_kernel<<<grid, 256, 0, stream>>>(Xv, Wvh, bv, Vb, MM, CC, CC);
    }
    {   // 4) WKV chunked scan + gate
        int t1 = NB * NCH * CC;
        wkv_pass1_kernel<<<(t1 + 255) / 256, 256, 0, stream>>>(Kb, Vb, td, Pe, Pev);
        int t2 = NB * CC;
        wkv_pass2_kernel<<<(t2 + 255) / 256, 256, 0, stream>>>(Pe, Pev);
        wkv_pass3_kernel<<<(t1 + 255) / 256, 256, 0, stream>>>(Rb, Kb, Vb, td, U, Pe, Pev, Gb);
    }
    {   // 5) output projection GEMM
        dim3 grid(MM / 128, CC / 128);
        gemm_f16_wmma_kernel<<<grid, 256, 0, stream>>>(Gb, Woh, bo, (float*)d_out, MM, CC, CC);
    }
}
